// KLinearFP8_41489384079599
// MI455X (gfx1250) — 8-bit matrix operand order repaired and checked by reading, compiled, not run on hardware
//
#include <hip/hip_runtime.h>
#include <cmath>


typedef int   v16i __attribute__((ext_vector_type(16)));
typedef float v8f  __attribute__((ext_vector_type(8)));

#define QBS    128
#define FP8MAX 448.0f

__device__ __forceinline__ int pack_fp8x4(float a, float b, float c, float d) {
#if __has_builtin(__builtin_amdgcn_cvt_pk_fp8_f32)
    int r = __builtin_amdgcn_cvt_pk_fp8_f32(a, b, 0, false);
    r = __builtin_amdgcn_cvt_pk_fp8_f32(c, d, r, true);
    return r;
#else
    int r = 0;
    asm volatile("v_cvt_pk_fp8_f32 %0, %1, %2" : "+v"(r) : "v"(a), "v"(b));
    asm volatile("v_cvt_pk_fp8_f32 %0, %1, %2 op_sel:[0,0,1]" : "+v"(r) : "v"(c), "v"(d));
    return r;
#endif
}

__device__ __forceinline__ void async_ld16(unsigned lds_off, const unsigned char* g) {
    asm volatile("global_load_async_to_lds_b128 %0, %1, off"
                 :: "v"(lds_off), "v"(g) : "memory");
}
#define WAIT_ASYNC_LE8() asm volatile("s_wait_asynccnt 0x8" ::: "memory")

__global__ __launch_bounds__(256) void quant_x_kernel(
    const float* __restrict__ x, unsigned char* __restrict__ xq,
    float* __restrict__ sxT, int M, int K)
{
    const int KB   = K / QBS;
    const int lane = threadIdx.x & 31;
    const long gw  = ((long)blockIdx.x * blockDim.x + threadIdx.x) >> 5;
    const int m  = (int)(gw / KB);
    const int kb = (int)(gw % KB);
    if (m >= M) return;

    const float4 v = *(const float4*)(x + (size_t)m * K + (size_t)kb * QBS + lane * 4);
    float amax = fmaxf(fmaxf(fabsf(v.x), fabsf(v.y)), fmaxf(fabsf(v.z), fabsf(v.w)));
    #pragma unroll
    for (int off = 16; off > 0; off >>= 1)
        amax = fmaxf(amax, __shfl_xor(amax, off, 32));

    const float s   = fmaxf(amax, 1e-20f) * (1.0f / FP8MAX);
    const float inv = 1.0f / s;

    const int q = pack_fp8x4(v.x * inv, v.y * inv, v.z * inv, v.w * inv);
    *(int*)(xq + (size_t)m * K + (size_t)kb * QBS + lane * 4) = q;
    if (lane == 0) sxT[(size_t)kb * M + m] = s;
}

__global__ __launch_bounds__(256) void quant_w_kernel(
    const float* __restrict__ w, unsigned char* __restrict__ wq, long total4)
{
    const long i = (long)blockIdx.x * blockDim.x + threadIdx.x;
    if (i >= total4) return;
    const float4 v = *(const float4*)(w + i * 4);
    *(int*)(wq + i * 4) = pack_fp8x4(v.x, v.y, v.z, v.w);
}

__global__ __launch_bounds__(256, 1) void gemm_fp8_kernel(
    const unsigned char* __restrict__ xq,
    const unsigned char* __restrict__ wq,
    const float* __restrict__ sxT,
    const float* __restrict__ ws,
    float* __restrict__ out,
    int M, int N, int K)
{
    __shared__ unsigned char smem[2][2][128 * 128];

    const int KB   = K / QBS;
    const int t    = threadIdx.x;
    const int lane = t & 31;
    const int wave = t >> 5;
    const int wm   = wave & 3;
    const int wn   = wave >> 2;
    const int m_wg = blockIdx.y * 128;
    const int n_wg = blockIdx.x * 128;
    const int half = (lane >> 4) & 1;
    const int l16  = lane & 15;
    const int nb   = blockIdx.x;

    const unsigned char* ga = xq + (size_t)m_wg * K;
    const unsigned char* gb = wq + (size_t)n_wg * K;

    auto stage = [&](int kb, int b) {
        #pragma unroll
        for (int i = 0; i < 4; ++i) {
            const int idx = i * 4096 + t * 16;
            const int row = idx >> 7, col = idx & 127;
            async_ld16((unsigned)(size_t)&smem[b][0][idx],
                       ga + (size_t)row * K + (size_t)kb * QBS + col);
        }
        #pragma unroll
        for (int i = 0; i < 4; ++i) {
            const int idx = i * 4096 + t * 16;
            const int row = idx >> 7, col = idx & 127;
            async_ld16((unsigned)(size_t)&smem[b][1][idx],
                       gb + (size_t)row * K + (size_t)kb * QBS + col);
        }
    };

    v8f acc[2][4];
    #pragma unroll
    for (int tm = 0; tm < 2; ++tm)
        #pragma unroll
        for (int tn = 0; tn < 4; ++tn)
            #pragma unroll
            for (int i = 0; i < 8; ++i) acc[tm][tn][i] = 0.0f;

    stage(0, 0);
    int buf = 0;

    for (int kb = 0; kb < KB; ++kb) {
        const int nkb = (kb + 1 < KB) ? kb + 1 : 0;
        stage(nkb, buf ^ 1);
        WAIT_ASYNC_LE8();
        __syncthreads();

        v16i afrag[2];
        #pragma unroll
        for (int tm = 0; tm < 2; ++tm) {
            const int arow = wm * 32 + tm * 16 + l16;
            #pragma unroll
            for (int h = 0; h < 2; ++h)
                #pragma unroll
                for (int j = 0; j < 4; ++j) {
                    const int2 q = *(const int2*)
                        &smem[buf][0][arow * 128 + h * 64 + j * 16 + half * 8];
                    afrag[tm][h * 8 + j * 2 + 0] = q.x;
                    afrag[tm][h * 8 + j * 2 + 1] = q.y;
                }
        }

        const float wsc = ws[(size_t)nb * KB + kb];
        float fa[2][8];
        #pragma unroll
        for (int tm = 0; tm < 2; ++tm) {
            const float* sp = sxT + (size_t)kb * M + m_wg + wm * 32 + tm * 16 + half * 8;
            const float4 s0 = *(const float4*)sp;
            const float4 s1 = *(const float4*)(sp + 4);
            fa[tm][0] = s0.x * wsc; fa[tm][1] = s0.y * wsc;
            fa[tm][2] = s0.z * wsc; fa[tm][3] = s0.w * wsc;
            fa[tm][4] = s1.x * wsc; fa[tm][5] = s1.y * wsc;
            fa[tm][6] = s1.z * wsc; fa[tm][7] = s1.w * wsc;
        }

        #pragma unroll
        for (int tn = 0; tn < 4; ++tn) {
            v16i bfrag;
            const int brow = wn * 64 + tn * 16 + l16;
            #pragma unroll
            for (int c = 0; c < 8; ++c) {
                const int2 q = *(const int2*)
                    &smem[buf][1][brow * 128 + (c >> 2) * 64 + (c & 3) * 16 + half * 8];
                bfrag[c * 2 + 0] = q.x; bfrag[c * 2 + 1] = q.y;
            }
            #pragma unroll
            for (int tm = 0; tm < 2; ++tm) {
                v8f z = {};
                v8f p = __builtin_amdgcn_wmma_f32_16x16x128_fp8_fp8(
                            afrag[tm], bfrag, (short)0, z, false, false);
                #pragma unroll
                for (int i = 0; i < 8; ++i)
                    acc[tm][tn][i] += p[i] * fa[tm][i];
            }
        }

        __syncthreads();
        buf ^= 1;
    }

    #pragma unroll
    for (int tm = 0; tm < 2; ++tm)
        #pragma unroll
        for (int tn = 0; tn < 4; ++tn)
            #pragma unroll
            for (int i = 0; i < 8; ++i) {
                const int m = m_wg + wm * 32 + tm * 16 + i + half * 8;
                const int n = n_wg + wn * 64 + tn * 16 + l16;
                out[(size_t)m * N + n] = acc[tm][tn][i];
            }
}

extern "C" void kernel_launch(void* const* d_in, const int* in_sizes, int n_in,
                              void* d_out, int out_size, void* d_ws, size_t ws_size,
                              hipStream_t stream) {
    const float* x   = (const float*)d_in[0];
    const float* w   = (const float*)d_in[1];
    const float* wsi = (const float*)d_in[2];

    const double a = (double)in_sizes[0], b = (double)in_sizes[1], c = (double)out_size;
    const int M = (int)llround(sqrt(a * c / b));
    const int K = (int)llround(a / (double)M);
    const int N = (int)llround(c / (double)M);
    const int KB = K / QBS;

    unsigned char* wq = (unsigned char*)d_ws;
    unsigned char* xq = wq + (size_t)N * K;
    float*        sxT = (float*)(xq + (size_t)M * K);

    {
        const long total4 = (long)N * K / 4;
        const int blocks = (int)((total4 + 255) / 256);
        quant_w_kernel<<<blocks, 256, 0, stream>>>(w, wq, total4);
    }
    {
        const long waves  = (long)M * KB;
        const int  blocks = (int)((waves + 7) / 8);
        quant_x_kernel<<<blocks, 256, 0, stream>>>(x, xq, sxT, M, K);
    }
    {
        dim3 grid(N / 128, M / 128);
        gemm_fp8_kernel<<<grid, 256, 0, stream>>>(xq, wq, sxT, wsi,
                                                  (float*)d_out, M, N, K);
    }
}
